// DecoderLayer_25151328485901
// MI455X (gfx1250) — matrix operand order repaired and checked by reading, compiled, not run on hardware
//
#include <hip/hip_runtime.h>
#include <math.h>

// ---------------- constants ----------------
#define DMODEL 512
#define NHEAD  8
#define HDIM   64
#define CELU_ALPHA 1.3f
#define LN_EPS 1e-5f

// ---------------- WMMA types ----------------
typedef __attribute__((ext_vector_type(16))) _Float16 v16h;
typedef __attribute__((ext_vector_type(4)))  _Float16 v4h;
typedef __attribute__((ext_vector_type(8)))  float    v8f;

// ---------------- wave reductions (wave32) ----------------
__device__ inline float wsum32(float v) {
#pragma unroll
    for (int o = 16; o > 0; o >>= 1) v += __shfl_xor(v, o, 32);
    return v;
}
__device__ inline float wmax32(float v) {
#pragma unroll
    for (int o = 16; o > 0; o >>= 1) v = fmaxf(v, __shfl_xor(v, o, 32));
    return v;
}

// ---------------- WMMA fragment loads from LDS ----------------
// A-matrix 16x32 f16 layout (ISA 7.12.2): lane<16 holds M=lane, K in {0..7,16..23};
// lane>=16 holds M=lane-16, K in {8..15,24..31}. Contiguous 8-half runs -> ds_load_b128.
__device__ inline v16h frag_a(const _Float16* As, int lda, int mbase, int lane) {
    v16h f;
    const _Float16* row = As + (mbase + (lane & 15)) * lda;
    int ko = (lane & 16) ? 8 : 0;
#pragma unroll
    for (int v = 0; v < 8; ++v) {
        int kb = ko + ((v & 4) ? 16 : 0) + ((v & 3) << 1);
        f[2 * v]     = row[kb];
        f[2 * v + 1] = row[kb + 1];
    }
    return f;
}
// B-matrix 32x16 f16: lane<16 holds N=lane&15, K in {0..7,16..23}; lane>=16 holds K in {8..15,24..31}.
// Bs stored transposed: Bs[n][k] -> two contiguous 8-half runs per lane.
__device__ inline v16h frag_b(const _Float16* Bs, int ldb, int nbase, int lane) {
    v16h f;
    const _Float16* row = Bs + (nbase + (lane & 15)) * ldb;
    int ko = (lane & 16) ? 8 : 0;
#pragma unroll
    for (int v = 0; v < 8; ++v) {
        int kb = ko + ((v & 4) ? 16 : 0) + ((v & 3) << 1);
        f[2 * v]     = row[kb];
        f[2 * v + 1] = row[kb + 1];
    }
    return f;
}

__device__ inline float apply_act(float v, int act) {
    if (act == 1) v = (v > 0.f) ? v : CELU_ALPHA * (expf(v * (1.0f / CELU_ALPHA)) - 1.f);
    else if (act == 2) v = fmaxf(v, 0.f);
    return v;
}

// C/D 16x16 f32 layout: VGPR r -> row = r + (lane>=16 ? 8 : 0), col = lane&15.
__device__ inline void store_tile(const v8f& c, float* __restrict__ Y,
                                  const float* __restrict__ bias,
                                  const float* __restrict__ res,
                                  int N, int rbase, int cbase, int lane, int act) {
    int mo = (lane & 16) ? 8 : 0;
    int n  = cbase + (lane & 15);
    float bv = bias ? bias[n] : 0.f;
#pragma unroll
    for (int r = 0; r < 8; ++r) {
        int row = rbase + mo + r;
        float v = c[r] + bv;
        v = apply_act(v, act);
        if (res) v += res[(size_t)row * N + n];
        Y[(size_t)row * N + n] = v;
    }
}

// ---------------- tiled WMMA GEMM: Y = act(X[RxK] @ W[KxN] + b) (+res) ----------------
// grid = (N/64, R/64), block = 128 (4 waves, 2x2, 32x32 per wave), double-buffered LDS.
#define LDT 40  // padded f16 row length (80B rows, 8B-aligned sub-offsets)
__global__ void __launch_bounds__(128)
gemm_wmma_kernel(const float* __restrict__ X, const float* __restrict__ W,
                 const float* __restrict__ bias, const float* __restrict__ res,
                 float* __restrict__ Y, int R, int K, int N, int act) {
    __shared__ __align__(16) _Float16 As[2][64 * LDT];  // [row][k]
    __shared__ __align__(16) _Float16 Bs[2][64 * LDT];  // [col][k] (W tile transposed)

    const int tid  = threadIdx.x;
    const int lane = tid & 31;
    const int wave = tid >> 5;
    const int wm   = wave >> 1;
    const int wn   = wave & 1;
    const int rowBase = blockIdx.y * 64;
    const int colBase = blockIdx.x * 64;

    // A staging: thread owns rows ar0+16i (i=0..3), cols ac..ac+3 of the 64x32 tile
    const int ar0 = tid >> 3;          // 0..15
    const int ac  = (tid & 7) * 4;     // 0..28
    const float* Xbase = X + (size_t)(rowBase + ar0) * K + ac;
    // B staging: thread owns a 4x4 patch at (k=bk0.., n=bn0..) of the 32x64 tile
    const int bk0 = (tid >> 4) * 4;    // 0..28
    const int bn0 = (tid & 15) * 4;    // 0..60
    const float* Wbase = W + (size_t)bk0 * N + colBase + bn0;

    float4 ra[4], rb[4];
    auto load_regs = [&](int k0) {
#pragma unroll
        for (int i = 0; i < 4; ++i)
            ra[i] = *(const float4*)(Xbase + (size_t)(16 * i) * K + k0);
#pragma unroll
        for (int i = 0; i < 4; ++i)
            rb[i] = *(const float4*)(Wbase + (size_t)(k0 + i) * N);
    };
    auto store_regs = [&](int nb) {
#pragma unroll
        for (int i = 0; i < 4; ++i) {
            v4h h = {(_Float16)ra[i].x, (_Float16)ra[i].y,
                     (_Float16)ra[i].z, (_Float16)ra[i].w};
            *(v4h*)&As[nb][(ar0 + 16 * i) * LDT + ac] = h;
        }
        float br[4][4] = {{rb[0].x, rb[0].y, rb[0].z, rb[0].w},
                          {rb[1].x, rb[1].y, rb[1].z, rb[1].w},
                          {rb[2].x, rb[2].y, rb[2].z, rb[2].w},
                          {rb[3].x, rb[3].y, rb[3].z, rb[3].w}};
#pragma unroll
        for (int j = 0; j < 4; ++j) {   // transpose 4x4 patch, packed b64 store
            v4h h = {(_Float16)br[0][j], (_Float16)br[1][j],
                     (_Float16)br[2][j], (_Float16)br[3][j]};
            *(v4h*)&Bs[nb][(bn0 + j) * LDT + bk0] = h;
        }
    };

    v8f c00 = {}, c01 = {}, c10 = {}, c11 = {};

    load_regs(0);
    store_regs(0);
    __syncthreads();

    int cur = 0;
    for (int k0 = 32; k0 <= K; k0 += 32) {
        const bool more = (k0 < K);
        if (more) load_regs(k0);   // global b128 loads overlap compute below

        v16h a0 = frag_a(As[cur], LDT, wm * 32,      lane);
        v16h a1 = frag_a(As[cur], LDT, wm * 32 + 16, lane);
        v16h b0 = frag_b(Bs[cur], LDT, wn * 32,      lane);
        v16h b1 = frag_b(Bs[cur], LDT, wn * 32 + 16, lane);

        c00 = __builtin_amdgcn_wmma_f32_16x16x32_f16(false, a0, false, b0, (short)0, c00, false, false);
        c01 = __builtin_amdgcn_wmma_f32_16x16x32_f16(false, a0, false, b1, (short)0, c01, false, false);
        c10 = __builtin_amdgcn_wmma_f32_16x16x32_f16(false, a1, false, b0, (short)0, c10, false, false);
        c11 = __builtin_amdgcn_wmma_f32_16x16x32_f16(false, a1, false, b1, (short)0, c11, false, false);

        if (more) store_regs(cur ^ 1);
        __syncthreads();
        cur ^= 1;
    }

    store_tile(c00, Y, bias, res, N, rowBase + wm * 32,      colBase + wn * 32,      lane, act);
    store_tile(c01, Y, bias, res, N, rowBase + wm * 32,      colBase + wn * 32 + 16, lane, act);
    store_tile(c10, Y, bias, res, N, rowBase + wm * 32 + 16, colBase + wn * 32,      lane, act);
    store_tile(c11, Y, bias, res, N, rowBase + wm * 32 + 16, colBase + wn * 32 + 16, lane, act);
}

// ---------------- GroupNorm over D=512 with 8 groups of 64 (in place) ----------------
__global__ void __launch_bounds__(256)
groupnorm_kernel(float* __restrict__ X, const float* __restrict__ g,
                 const float* __restrict__ b, int R) {
    int row = blockIdx.x;
    int wave = threadIdx.x >> 5, lane = threadIdx.x & 31;
    size_t base = (size_t)row * DMODEL + wave * HDIM;
    float x0 = X[base + lane], x1 = X[base + lane + 32];
    float s  = wsum32(x0 + x1);
    float sq = wsum32(x0 * x0 + x1 * x1);
    float mu = s * (1.0f / HDIM);
    float var = sq * (1.0f / HDIM) - mu * mu;
    float rs = rsqrtf(var + LN_EPS);
    int c0 = wave * HDIM + lane, c1 = c0 + 32;
    X[base + lane]      = (x0 - mu) * rs * g[c0] + b[c0];
    X[base + lane + 32] = (x1 - mu) * rs * g[c1] + b[c1];
}

// ---------------- LayerNorm over D=512, optional residual ----------------
__global__ void __launch_bounds__(256)
layernorm_kernel(const float* __restrict__ A, const float* __restrict__ Badd,
                 const float* __restrict__ g, const float* __restrict__ be,
                 float* __restrict__ Y, int R) {
    __shared__ float red[16];
    int row = blockIdx.x, tid = threadIdx.x;
    size_t base = (size_t)row * DMODEL;
    float x0 = A[base + tid]       + (Badd ? Badd[base + tid]       : 0.f);
    float x1 = A[base + tid + 256] + (Badd ? Badd[base + tid + 256] : 0.f);
    int wave = tid >> 5, lane = tid & 31;
    float ps = wsum32(x0 + x1);
    float pq = wsum32(x0 * x0 + x1 * x1);
    if (lane == 0) { red[wave] = ps; red[8 + wave] = pq; }
    __syncthreads();
    float S = 0.f, Q = 0.f;
#pragma unroll
    for (int i = 0; i < 8; ++i) { S += red[i]; Q += red[8 + i]; }
    float mu = S * (1.0f / DMODEL);
    float var = Q * (1.0f / DMODEL) - mu * mu;
    float rs = rsqrtf(var + LN_EPS);
    Y[base + tid]       = (x0 - mu) * rs * g[tid]       + be[tid];
    Y[base + tid + 256] = (x1 - mu) * rs * g[tid + 256] + be[tid + 256];
}

// ---------------- elementwise add ----------------
__global__ void __launch_bounds__(256)
add_kernel(const float* __restrict__ A, const float* __restrict__ B,
           float* __restrict__ C, int n) {
    int i = blockIdx.x * 256 + threadIdx.x;
    if (i < n) C[i] = A[i] + B[i];
}

// ---------------- concat rows: out[r] = [A[r] , B[r]] ----------------
__global__ void __launch_bounds__(256)
concat_kernel(const float* __restrict__ A, const float* __restrict__ B,
              float* __restrict__ O, int R) {
    int row = blockIdx.x, tid = threadIdx.x;
#pragma unroll
    for (int j = 0; j < 4; ++j) {
        int col = tid + j * 256;
        float v = (col < DMODEL) ? A[(size_t)row * DMODEL + col]
                                 : B[(size_t)row * DMODEL + col - DMODEL];
        O[(size_t)row * 1024 + col] = v;
    }
}

// ---------------- low-rank scatt core: one wave per (b,h,q) ----------------
__global__ void __launch_bounds__(256)
scatt_kernel(const float* __restrict__ q, const float* __restrict__ k,
             const float* __restrict__ v1, const float* __restrict__ v2,
             const float* __restrict__ mask,
             const float* __restrict__ basic_w, const float* __restrict__ basic_b,
             const float* __restrict__ last_w,  const float* __restrict__ last_b,
             const float* __restrict__ last2_w, const float* __restrict__ last2_b,
             float* __restrict__ out, int Hn, int Lq, int Lk) {
    __shared__ float bw[64 * 32];     // basic_w staged (hot operand)
    __shared__ float qv[8][64];
    __shared__ float lg[8][160];
    __shared__ float pl[8][32];

    const int wv = threadIdx.x >> 5, lane = threadIdx.x & 31;
#pragma unroll
    for (int i = threadIdx.x; i < 64 * 32; i += 256) bw[i] = basic_w[i];

    const int gidx = blockIdx.x * 8 + wv;
    const int b  = gidx / (Hn * Lq);
    const int r  = gidx % (Hn * Lq);
    const int h  = r / Lq;
    const int qi = r % Lq;

    const size_t qbase = ((size_t)b * Lq + qi) * DMODEL + h * HDIM;
    qv[wv][lane]      = q[qbase + lane];
    qv[wv][lane + 32] = q[qbase + lane + 32];
    __syncthreads();

    const float lw = last_w[lane];
    const float bb = basic_b[lane];
    const float* mrow = mask + ((size_t)b * Lq + qi) * Lk;
    float pool = 0.f, msum = 0.f;

    for (int kk = 0; kk < Lk; ++kk) {
        float mv = mrow[kk];
        const float4* kr4 = (const float4*)(k + ((size_t)b * Lk + kk) * DMODEL + h * HDIM);
        float acc = bb;
#pragma unroll
        for (int d4 = 0; d4 < 16; ++d4) {
            float4 kv = kr4[d4];                 // broadcast b128
            int d = d4 * 4;
            acc = fmaf(qv[wv][d + 0] * kv.x, bw[(d + 0) * 32 + lane], acc);
            acc = fmaf(qv[wv][d + 1] * kv.y, bw[(d + 1) * 32 + lane], acc);
            acc = fmaf(qv[wv][d + 2] * kv.z, bw[(d + 2) * 32 + lane], acc);
            acc = fmaf(qv[wv][d + 3] * kv.w, bw[(d + 3) * 32 + lane], acc);
        }
        float hm = fmaxf(acc, 0.f);              // relu(hmid)
        pool += hm * mv;
        msum += mv;
        float lp = wsum32(hm * lw);
        if (lane == 0) lg[wv][kk] = (mv == 0.f) ? -1e9f : (lp + last_b[0]);
    }

    pool /= msum;
    pl[wv][lane] = pool;
    __syncthreads();

    // channel gate: sigmoid(pool @ last2_w + last2_b)
    float s0 = last2_b[lane], s1 = last2_b[lane + 32];
#pragma unroll
    for (int m = 0; m < 32; ++m) {
        float pm = pl[wv][m];
        s0 = fmaf(pm, last2_w[m * 64 + lane], s0);
        s1 = fmaf(pm, last2_w[m * 64 + lane + 32], s1);
    }
    float ach0 = 1.f / (1.f + expf(-s0));
    float ach1 = 1.f / (1.f + expf(-s1));

    // softmax over kk
    float mx = -1e30f;
    for (int kk = lane; kk < Lk; kk += 32) mx = fmaxf(mx, lg[wv][kk]);
    mx = wmax32(mx);
    float se = 0.f;
    for (int kk = lane; kk < Lk; kk += 32) {
        float e = expf(lg[wv][kk] - mx);
        lg[wv][kk] = e;
        se += e;
    }
    se = wsum32(se);
    __syncthreads();

    // v2 aggregation
    float va0 = 0.f, va1 = 0.f;
    for (int kk = 0; kk < Lk; ++kk) {
        float a = lg[wv][kk];
        const float* vr = v2 + ((size_t)b * Lk + kk) * DMODEL + h * HDIM;
        va0 = fmaf(a, vr[lane], va0);
        va1 = fmaf(a, vr[lane + 32], va1);
    }
    float inv = 1.f / se;
    out[qbase + lane]      = v1[qbase + lane]      * (va0 * inv) * ach0;
    out[qbase + lane + 32] = v1[qbase + lane + 32] * (va1 * inv) * ach1;
}

// ---------------- host orchestration ----------------
static inline void run_gemm(const float* X, const float* W, const float* bias,
                            const float* res, float* Y, int R, int K, int N,
                            int act, hipStream_t s) {
    dim3 grid(N / 64, R / 64);
    gemm_wmma_kernel<<<grid, 128, 0, s>>>(X, W, bias, res, Y, R, K, N, act);
}

extern "C" void kernel_launch(void* const* d_in, const int* in_sizes, int n_in,
                              void* d_out, int out_size, void* d_ws, size_t ws_size,
                              hipStream_t stream) {
    const int B = 64, L = 20, M = 144;
    const int RX = B * L;   // 1280
    const int RE = B * M;   // 9216

    const float* gx   = (const float*)d_in[0];
    const float* x    = (const float*)d_in[1];
    const float* enc  = (const float*)d_in[2];
    const float* amsk = (const float*)d_in[3];
    const float* smsk = (const float*)d_in[4];

    const int WB = 5, CB = 27;   // params: word 5.., cross 27.., ln 49.., tail 57..62
    auto P = [&](int i) { return (const float*)d_in[i]; };

    float* w = (float*)d_ws;
    size_t o = 0;
    auto alloc = [&](size_t n) { float* p = w + o; o += n; return p; };
    float* big0 = alloc((size_t)RE * DMODEL);   // cross k
    float* big1 = alloc((size_t)RE * DMODEL);   // cross v2
    float* t0 = alloc((size_t)RX * DMODEL);
    float* t1 = alloc((size_t)RX * DMODEL);
    float* t2 = alloc((size_t)RX * DMODEL);
    float* t3 = alloc((size_t)RX * DMODEL);
    float* t4 = alloc((size_t)RX * DMODEL);
    float* t5 = alloc((size_t)RX * DMODEL);
    float* wide = alloc((size_t)RX * 2048);     // concat (1024) / ffh (2048)
    (void)ws_size; (void)in_sizes; (void)n_in; (void)out_size;

    float* gx_out = (float*)d_out;
    float* x_out  = (float*)d_out + (size_t)RX * DMODEL;

    auto proj = [&](const float* X_, int rows, int base, int nm, float* out_) {
        int wi = base + nm * 4;
        run_gemm(X_, P(wi), P(wi + 1), nullptr, out_, rows, DMODEL, DMODEL, /*celu*/1, stream);
        groupnorm_kernel<<<rows, 256, 0, stream>>>(out_, P(wi + 2), P(wi + 3), rows);
    };

    // -------- word scatt --------
    proj(gx, RX, WB, 0, t0);   // q
    proj(x,  RX, WB, 1, t1);   // k
    proj(gx, RX, WB, 2, t2);   // v1
    proj(x,  RX, WB, 3, t3);   // v2
    scatt_kernel<<<(B * NHEAD * L) / 8, 256, 0, stream>>>(
        t0, t1, t2, t3, smsk,
        P(WB + 16), P(WB + 17), P(WB + 18), P(WB + 19), P(WB + 20), P(WB + 21),
        t4, NHEAD, L, L);

    add_kernel<<<(RX * DMODEL + 255) / 256, 256, 0, stream>>>(x, t4, t5, RX * DMODEL);
    layernorm_kernel<<<RX, 256, 0, stream>>>(t5, nullptr, P(49), P(50), t0, RX);

    // -------- cross scatt --------
    proj(t0,  RX, CB, 0, t1);   // q
    proj(enc, RE, CB, 1, big0); // k
    proj(t0,  RX, CB, 2, t2);   // v1
    proj(enc, RE, CB, 3, big1); // v2
    scatt_kernel<<<(B * NHEAD * L) / 8, 256, 0, stream>>>(
        t1, big0, t2, big1, amsk,
        P(CB + 16), P(CB + 17), P(CB + 18), P(CB + 19), P(CB + 20), P(CB + 21),
        t3, NHEAD, L, M);

    layernorm_kernel<<<RX, 256, 0, stream>>>(t5, t3, P(51), P(52), gx_out, RX);

    // bifeat
    concat_kernel<<<RX, 256, 0, stream>>>(gx_out, x, wide, RX);
    run_gemm(wide, P(57), P(58), x, t4, RX, 2 * DMODEL, DMODEL, /*celu*/1, stream);
    layernorm_kernel<<<RX, 256, 0, stream>>>(t4, nullptr, P(53), P(54), t0, RX);

    // FF
    run_gemm(t0, P(59), P(60), nullptr, wide, RX, DMODEL, 4 * DMODEL, /*relu*/2, stream);
    run_gemm(wide, P(61), P(62), nullptr, t1, RX, 4 * DMODEL, DMODEL, /*none*/0, stream);
    layernorm_kernel<<<RX, 256, 0, stream>>>(t0, t1, P(55), P(56), x_out, RX);
}
